// DeblurNet_35923106464293
// MI455X (gfx1250) — matrix operand order repaired and checked by reading, not compiled, not run on hardware
//
#include <hip/hip_runtime.h>

typedef __attribute__((ext_vector_type(16))) _Float16 v16h;
typedef __attribute__((ext_vector_type(8)))  float    v8f;

// ---------------------------------------------------------------------------
// Weight pre-pack: f32 OIHW -> f16 tiles in WMMA A-operand layout:
//   dst[((mt*KC + kc)*32 + lane)*16 + e],  e = 2*j+h,
//   m = mt*16 + (lane&15),
//   k = kc*32 + (j<4 ? 2j : 16+2(j-4)) + (lane>=16 ? 8 : 0) + h
// Cout/Ktot tails baked in as zeros.
// ---------------------------------------------------------------------------
__global__ void pack_w_kernel(const float* __restrict__ w, _Float16* __restrict__ dst,
                              int Cout, int Ktot)
{
    int KC = (Ktot + 31) >> 5;
    int MT = (Cout + 15) >> 4;
    int total = MT * KC * 512;
    int idx = blockIdx.x * blockDim.x + threadIdx.x;
    if (idx >= total) return;
    int e = idx & 15; int t = idx >> 4;
    int lane = t & 31; t >>= 5;
    int kc = t % KC;   int mt = t / KC;
    int j = e >> 1, h = e & 1;
    int m = mt * 16 + (lane & 15);
    int k = kc * 32 + (j < 4 ? 2 * j : 16 + 2 * (j - 4)) + ((lane >> 4) ? 8 : 0) + h;
    float v = (m < Cout && k < Ktot) ? w[(size_t)m * Ktot + k] : 0.f;
    dst[idx] = (_Float16)v;
}

// ---------------------------------------------------------------------------
// Input staging: dense NCHW -> zero-padded (Cin+1 channels, channel Cin = 0).
// Removes every bounds check from the WMMA gather loop.
// ---------------------------------------------------------------------------
__global__ void pad_stage_kernel(const float* __restrict__ x, float* __restrict__ xp,
                                 int Bn, int Cin, int Hin, int Win, int pad)
{
    int Hp = Hin + 2 * pad, Wp = Win + 2 * pad, Cp = Cin + 1;
    int total = Bn * Cp * Hp * Wp;
    int idx = blockIdx.x * blockDim.x + threadIdx.x;
    if (idx >= total) return;
    int wq = idx % Wp; int t = idx / Wp;
    int hq = t % Hp;   t /= Hp;
    int c  = t % Cp;   int b = t / Cp;
    int hi = hq - pad, wi = wq - pad;
    float v = 0.f;
    if (c < Cin && hi >= 0 && hi < Hin && wi >= 0 && wi < Win)
        v = x[((size_t)(b * Cin + c) * Hin + hi) * Win + wi];
    xp[idx] = v;
}

// ---------------------------------------------------------------------------
// Implicit-GEMM conv via V_WMMA_F32_16X16X32_F16 on the padded input.
// One wave32 computes a 16(Cout) x 64(pixels) D tile; per K-chunk: 4 WMMAs.
// A operand is software-pipelined (next chunk loaded during current compute)
// and prefetched two chunks ahead (global_prefetch_b8, speculative).
// Address = xp + tileoff[t] (K-invariant) + chanoff (tile-invariant).
// act: 0 none, 1 leaky_relu(0.1), 2 prelu(alpha_p[0]); res -> fused add.
// ---------------------------------------------------------------------------
template<int K, int STRIDE>
__global__ __launch_bounds__(128)
void conv2d_wmma(const float* __restrict__ xp, const _Float16* __restrict__ wpk,
                 const float* __restrict__ bias, const float* __restrict__ res,
                 const float* __restrict__ alpha_p, float* __restrict__ y,
                 int Bn, int Cin, int Hp, int Wp, int Cout, int Ho, int Wo, int act)
{
    const int lane = threadIdx.x & 31;
    const int wv   = threadIdx.x >> 5;
    const int Npix = Bn * Ho * Wo;
    const int pixBase = (blockIdx.x * 4 + wv) * 64;
    if (pixBase >= Npix) return;                       // wave-uniform
    const int coutBase = blockIdx.y * 16;
    constexpr int KK = K * K;
    const int Ktot = Cin * KK;
    const int KC   = (Ktot + 31) >> 5;
    const int Cp   = Cin + 1;

    // Per-wave pixel decomposition: each 16-pixel sub-tile lies in one row.
    int bt[4], hot[4], wot[4], tileoff[4];
    #pragma unroll
    for (int t = 0; t < 4; ++t) {
        int p0 = pixBase + t * 16;
        int b  = p0 / (Ho * Wo); int rp = p0 - b * (Ho * Wo);
        bt[t] = b; hot[t] = rp / Wo; wot[t] = rp - hot[t] * Wo;
        tileoff[t] = (b * Cp * Hp + hot[t] * STRIDE) * Wp + wot[t] * STRIDE;
    }

    const _Float16* wtile = wpk + ((size_t)blockIdx.y * KC * 32 + lane) * 16;

    v8f acc[4] = {v8f{}, v8f{}, v8f{}, v8f{}};
    v16h aM = *(const v16h*)(wtile);                   // chunk 0 A tile
    for (int kc = 0; kc < KC; ++kc) {
        // Software-pipelined A: fetch next chunk (clamped -> always valid).
        int kcn = min(kc + 1, KC - 1);
        v16h aN = *(const v16h*)(wtile + (size_t)kcn * 512);
        // Speculative prefetch two chunks ahead (global_prefetch_b8).
        __builtin_prefetch(wtile + (size_t)(kc + 2) * 512, 0, 0);

        const int nB = lane & 15, hB = lane >> 4;

        // ---- hoist ALL sub-tile loads ahead of converts / WMMAs ----
        float tv[4][16];
        #pragma unroll
        for (int s = 0; s < 16; ++s) {
            int kk = kc * 32 + (s < 8 ? 8 * hB + s : 16 + 8 * hB + (s - 8));
            int q  = kk / KK;                              // constant-divisor
            int rr = kk - q * KK;
            int kh = rr / K, kw = rr - (rr / K) * K;
            int ci = min(q, Cin);                          // tail rows -> zero channel
            int chanoff = (ci * Hp + kh) * Wp + kw;        // tile-invariant
            #pragma unroll
            for (int t = 0; t < 4; ++t) tv[t][s] = xp[tileoff[t] + chanoff + nB * STRIDE];
        }
        v16h bM[4];
        #pragma unroll
        for (int t = 0; t < 4; ++t)
            #pragma unroll
            for (int n = 0; n < 16; ++n) bM[t][n] = (_Float16)tv[t][n];
        #pragma unroll
        for (int t = 0; t < 4; ++t)
            acc[t] = __builtin_amdgcn_wmma_f32_16x16x32_f16(false, aM, false, bM[t],
                                                            (short)0, acc[t], false, false);
        aM = aN;
    }

    // Epilogue: bias + residual + activation.  D: n = lane&15, m = r + 8*(lane>=16).
    float alpha = (act == 2) ? alpha_p[0] : 0.1f;
    const int n    = lane & 15;
    const int mofs = (lane >> 4) ? 8 : 0;
    const int chw  = Ho * Wo;
    #pragma unroll
    for (int t = 0; t < 4; ++t) {
        int base = bt[t] * Cout * chw + hot[t] * Wo + (wot[t] + n);
        #pragma unroll
        for (int r = 0; r < 8; ++r) {
            int mm = coutBase + r + mofs;
            if (mm < Cout) {
                int oi = base + mm * chw;
                float v = acc[t][r] + bias[mm];
                if (res) v += res[oi];
                if (act == 1)      v = (v >= 0.f) ? v : 0.1f * v;
                else if (act == 2) v = (v >= 0.f) ? v : alpha * v;
                y[oi] = v;
            }
        }
    }
}

// ---------------------------------------------------------------------------
// Per-(b,c) max over HxW, clamped at 0.05.
// ---------------------------------------------------------------------------
__global__ void hm_max_kernel(const float* __restrict__ hm, float* __restrict__ mx, int HW)
{
    __shared__ float red[256];
    const float* p = hm + (size_t)blockIdx.x * HW;
    float m = -3.4e38f;
    for (int i = threadIdx.x; i < HW; i += 256) m = fmaxf(m, p[i]);
    red[threadIdx.x] = m;
    __syncthreads();
    for (int s = 128; s > 0; s >>= 1) {
        if ((int)threadIdx.x < s) red[threadIdx.x] = fmaxf(red[threadIdx.x], red[threadIdx.x + s]);
        __syncthreads();
    }
    if (threadIdx.x == 0) mx[blockIdx.x] = fmaxf(red[0], 0.05f);
}

__global__ void hm_merge5_kernel(const float* __restrict__ hm, const float* __restrict__ mx,
                                 float* __restrict__ hm5, int Bn, int HW)
{
    int idx = blockIdx.x * blockDim.x + threadIdx.x;
    if (idx >= Bn * HW) return;
    int b = idx / HW, pix = idx - b * HW;
    const float* hb = hm + (size_t)b * 68 * HW + pix;
    const float* mb = mx + b * 68;
    float g[5] = {0.f, 0.f, 0.f, 0.f, 0.f};
    for (int c = 36; c < 42; ++c) g[0] += hb[(size_t)c * HW] / mb[c];
    for (int c = 42; c < 48; ++c) g[1] += hb[(size_t)c * HW] / mb[c];
    for (int c = 27; c < 36; ++c) g[2] += hb[(size_t)c * HW] / mb[c];
    for (int c = 48; c < 68; ++c) g[3] += hb[(size_t)c * HW] / mb[c];
    for (int c = 0;  c < 27; ++c) g[4] += hb[(size_t)c * HW] / mb[c];
    for (int gg = 0; gg < 5; ++gg) hm5[((size_t)b * 5 + gg) * HW + pix] = g[gg];
}

__global__ void fusion_apply_kernel(const float* __restrict__ x160, const float* __restrict__ hm5,
                                    float* __restrict__ out, int Bn, int C, int HW)
{
    int idx = blockIdx.x * blockDim.x + threadIdx.x;
    if (idx >= Bn * C * HW) return;
    int pix = idx % HW; int t = idx / HW; int c = t % C; int b = t / C;
    float a[5], mxv = -3.4e38f;
    for (int g = 0; g < 5; ++g) { a[g] = hm5[((size_t)b * 5 + g) * HW + pix]; mxv = fmaxf(mxv, a[g]); }
    float den = 0.f;
    for (int g = 0; g < 5; ++g) { a[g] = expf(a[g] - mxv); den += a[g]; }
    float s = 0.f;
    for (int g = 0; g < 5; ++g)
        s += x160[((size_t)(b * 5 + g) * C + c) * HW + pix] * a[g];
    out[((size_t)b * C + c) * HW + pix] = s / den;
}

// Spatially-variant 5x5 per-pixel conv with edge padding (FAC).
__global__ void fac_conv_kernel(const float* __restrict__ feat, const float* __restrict__ kern,
                                float* __restrict__ out, int Bn, int C, int H, int W)
{
    int idx = blockIdx.x * blockDim.x + threadIdx.x;
    if (idx >= Bn * C * H * W) return;
    int w0 = idx % W; int t = idx / W; int h0 = t % H; t /= H; int c = t % C; int b = t / C;
    float s = 0.f;
    #pragma unroll
    for (int q = 0; q < 25; ++q) {
        int dh = q / 5 - 2, dw = q % 5 - 2;
        int hh = min(max(h0 + dh, 0), H - 1);
        int ww = min(max(w0 + dw, 0), W - 1);
        s += feat[((size_t)(b * C + c) * H + hh) * W + ww]
           * kern[((size_t)(b * C * 25 + c * 25 + q) * H + h0) * W + w0];
    }
    out[((size_t)(b * C + c) * H + h0) * W + w0] = s;
}

// 4x4 stride-2 transposed conv (lhs-dilated conv, pad 2) + leaky_relu(0.1).
__global__ void upconv_kernel(const float* __restrict__ x, const float* __restrict__ w,
                              const float* __restrict__ bias, float* __restrict__ y,
                              int Bn, int Cin, int Hi, int Wi, int Cout)
{
    int Ho = 2 * Hi, Wo = 2 * Wi;
    int idx = blockIdx.x * blockDim.x + threadIdx.x;
    if (idx >= Bn * Cout * Ho * Wo) return;
    int wo = idx % Wo; int t = idx / Wo; int ho = t % Ho; t /= Ho; int co = t % Cout; int b = t / Cout;
    float s = bias[co];
    #pragma unroll
    for (int kh = 0; kh < 4; ++kh) {
        int qh = ho + kh - 2;
        if (qh < 0 || qh > 2 * Hi - 2 || (qh & 1)) continue;
        int hi = qh >> 1;
        #pragma unroll
        for (int kw = 0; kw < 4; ++kw) {
            int qw = wo + kw - 2;
            if (qw < 0 || qw > 2 * Wi - 2 || (qw & 1)) continue;
            int wi = qw >> 1;
            const float* xptr = x + ((size_t)b * Cin * Hi + hi) * Wi + wi;
            const float* wptr = w + (size_t)co * Cin * 16 + kh * 4 + kw;
            for (int ci = 0; ci < Cin; ++ci)
                s += xptr[(size_t)ci * Hi * Wi] * wptr[(size_t)ci * 16];
        }
    }
    y[((size_t)(b * Cout + co) * Ho + ho) * Wo + wo] = (s >= 0.f) ? s : 0.1f * s;
}

__global__ void chan_copy_kernel(const float* __restrict__ src, float* __restrict__ dst,
                                 int Bn, int Csrc, int Cdst, int coff, int HW)
{
    int idx = blockIdx.x * blockDim.x + threadIdx.x;
    if (idx >= Bn * Csrc * HW) return;
    int pix = idx % HW; int t = idx / HW; int c = t % Csrc; int b = t / Csrc;
    dst[((size_t)(b * Cdst + coff + c)) * (size_t)HW + pix] =
        src[((size_t)(b * Csrc + c)) * (size_t)HW + pix];
}

// ---------------------------------------------------------------------------
// Host orchestration.  Param leaf indices (setup_inputs() insertion order,
// recursive; conv {'w','b'}, res {'w1','b1','w2','b2'}, fac {c1,r1,r2,c2}):
//   0:conv1_1  2:conv1_2  6:conv1_3  10:conv2_1 12:conv2_2 16:conv2_3
//   20:conv3_1 22:conv3_2 26:conv3_3 30:fscale1_1 32:fscale1_2 36:fscale1_3
//   40:kconv1_1 42:kconv1_2 46:kconv1_3 50:kconv2_1 52:kconv2_2 56:kconv2_3
//   60:kconv3_1 62:kconv3_2 66:kconv3_3 70:fac_warp(12) 82:kconv4
//   84:fac_deblur(12) 96:fea1 98:upconv2_u 100:upconv2_2 104:upconv2_1
//   108:upconv1_u 110:upconv1_2 114:upconv1_1 118:img_prd 120:HG2(c1,c2,c3,out)
//   128:fusion conv_in 130:fusion blocks(7x4) 158:compress conv 160:a
//   161..165: img_blur, last_img_blur, output_last_img, output_last_fea,
//             fusion_last_feature
// ---------------------------------------------------------------------------
extern "C" void kernel_launch(void* const* d_in, const int* in_sizes, int n_in,
                              void* d_out, int out_size, void* d_ws, size_t ws_size,
                              hipStream_t stream)
{
    (void)in_sizes; (void)n_in; (void)out_size; (void)ws_size;
    auto pf = [&](int i) { return (const float*)d_in[i]; };

    const int Bn = 2, H = 128, Wd = 128;
    const int HW = H * Wd, HW2 = 64 * 64, HW4 = 32 * 32;

    float* ws = (float*)d_ws;
    size_t off = 0;
    auto alloc = [&](size_t n) { float* p = ws + off; off += n; return p; };

    float* t128     = alloc((size_t)Bn * 128 * HW);
    float* t256a    = alloc((size_t)Bn * 256 * HW);
    float* t256b    = alloc((size_t)Bn * 256 * HW);
    float* kwarp    = t256a;   // 3200ch@32x32 fits; lifetime disjoint from HG2 temps
    float* kdeb     = t256b;
    float* xpad     = alloc((size_t)Bn * 257 * 130 * 130);  // staging (max Cin=256,pad=1)
    float* hm68     = alloc((size_t)Bn * 68 * HW);
    float* mx       = alloc((size_t)Bn * 68 + 8);
    float* hm5      = alloc((size_t)Bn * 5 * HW);
    float* merge    = alloc((size_t)Bn * 12 * HW);
    float* c1buf    = alloc((size_t)Bn * 32 * HW);
    float* k1buf    = alloc((size_t)Bn * 32 * HW);
    float* k2buf    = alloc((size_t)Bn * 64 * HW2);
    float* k3buf    = alloc((size_t)Bn * 128 * HW4);
    float* kc4buf   = alloc((size_t)Bn * 128 * HW4);
    float* cat256   = alloc((size_t)Bn * 256 * HW4);
    float* fach     = alloc((size_t)Bn * 128 * HW4);
    float* fused160 = alloc((size_t)Bn * 160 * HW);
    float* tmpbig   = alloc((size_t)Bn * 160 * HW);   // shared res-block temp
    float* fused32  = alloc((size_t)Bn * 32 * HW);
    float* featbuf  = alloc((size_t)Bn * 64 * HW4);
    float* c2buf    = alloc((size_t)Bn * 64 * HW2);
    float* c3buf    = alloc((size_t)Bn * 128 * HW4);
    float* c3k      = alloc((size_t)Bn * 128 * HW4);
    float* olff     = alloc((size_t)Bn * 128 * HW4);
    float* cakbuf   = alloc((size_t)Bn * 128 * HW4);
    float* cat320   = alloc((size_t)Bn * 320 * HW4);
    float* conv3buf = alloc((size_t)Bn * 256 * HW4);
    float* up2buf   = alloc((size_t)Bn * 64 * HW2);
    float* up1buf   = alloc((size_t)Bn * 32 * HW);

    // ---- f16 packed-weight arena (32B aligned) ----
    off = (off + 7) & ~(size_t)7;
    _Float16* wpack = (_Float16*)(ws + off);
    struct WPE { int wi, cout, ktot; };
    static const WPE wtab[] = {
        {0,32,27},{2,32,288},{4,32,288},{6,32,288},{8,32,288},
        {10,64,288},{12,64,576},{14,64,576},{16,64,576},{18,64,576},
        {20,128,576},{22,128,1152},{24,128,1152},{26,128,1152},{28,128,1152},
        {30,64,288},{32,64,576},{34,64,576},{36,64,576},{38,64,576},
        {40,32,108},{42,32,288},{44,32,288},{46,32,288},{48,32,288},
        {50,64,288},{52,64,576},{54,64,576},{56,64,576},{58,64,576},
        {60,128,576},{62,128,1152},{64,128,1152},{66,128,1152},{68,128,1152},
        {70,128,1152},{72,128,1152},{74,128,1152},{76,128,1152},{78,128,1152},{80,3200,128},
        {82,128,3200},
        {84,128,2304},{86,128,1152},{88,128,1152},{90,128,1152},{92,128,1152},{94,3200,128},
        {96,128,2304},
        {118,3,288},
        {120,128,27},{122,256,1152},{124,256,2304},{126,68,2304},
        {128,160,288},
        {130,160,1440},{132,160,1440},{134,160,1440},{136,160,1440},{138,160,1440},
        {140,160,1440},{142,160,1440},{144,160,1440},{146,160,1440},{148,160,1440},
        {150,160,1440},{152,160,1440},{154,160,1440},{156,160,1440},
        {158,256,320}
    };
    const _Float16* wptr[166] = {};
    {
        size_t hoff = 0;
        for (const WPE& e : wtab) {
            int MT = (e.cout + 15) / 16, KC = (e.ktot + 31) / 32;
            size_t nh = (size_t)MT * KC * 512;
            wptr[e.wi] = wpack + hoff;
            int total = (int)nh;
            pack_w_kernel<<<(total + 255) / 256, 256, 0, stream>>>(pf(e.wi), wpack + hoff,
                                                                   e.cout, e.ktot);
            hoff += nh;
        }
    }

    float* outp      = (float*)d_out;
    float* out_imgs  = outp;                                    // [2,B,3,H,W]
    float* out_conv3 = outp + (size_t)2 * Bn * 3 * HW;          // [B,256,32,32]
    float* out_hms   = out_conv3 + (size_t)Bn * 256 * HW4;      // [2,B,68,H,W]
    float* out_feat  = out_hms + (size_t)2 * Bn * 68 * HW;      // [B,64,32,32]

    auto conv = [&](const float* x, int wi, int bi, const float* res, float* y,
                    int Cin, int Hin, int Win, int Cout, int K, int stride, int pad, int act) {
        int Hp = Hin + 2 * pad, Wp = Win + 2 * pad;
        int Ho = (Hin + 2 * pad - K) / stride + 1;
        int Wo = (Win + 2 * pad - K) / stride + 1;
        { int total = Bn * (Cin + 1) * Hp * Wp;
          pad_stage_kernel<<<(total + 255) / 256, 256, 0, stream>>>(x, xpad, Bn, Cin, Hin, Win, pad); }
        int Npix = Bn * Ho * Wo;
        dim3 grid((Npix + 255) / 256, (Cout + 15) / 16);
        if (K == 3 && stride == 1)
            conv2d_wmma<3,1><<<grid, dim3(128), 0, stream>>>(xpad, wptr[wi], pf(bi), res, pf(160), y,
                Bn, Cin, Hp, Wp, Cout, Ho, Wo, act);
        else if (K == 3 && stride == 2)
            conv2d_wmma<3,2><<<grid, dim3(128), 0, stream>>>(xpad, wptr[wi], pf(bi), res, pf(160), y,
                Bn, Cin, Hp, Wp, Cout, Ho, Wo, act);
        else if (K == 3 && stride == 4)
            conv2d_wmma<3,4><<<grid, dim3(128), 0, stream>>>(xpad, wptr[wi], pf(bi), res, pf(160), y,
                Bn, Cin, Hp, Wp, Cout, Ho, Wo, act);
        else
            conv2d_wmma<1,1><<<grid, dim3(128), 0, stream>>>(xpad, wptr[wi], pf(bi), res, pf(160), y,
                Bn, Cin, Hp, Wp, Cout, Ho, Wo, act);
    };
    auto resblock = [&](float* x, int q0, int C, int Hh, int Ww) {
        conv(x, q0, q0 + 1, nullptr, tmpbig, C, Hh, Ww, C, 3, 1, 1, 1);
        conv(tmpbig, q0 + 2, q0 + 3, x, x, C, Hh, Ww, C, 3, 1, 1, 0);
    };
    auto copyCh = [&](const float* src, float* dst, int Csrc, int Cdst, int coff, int hw) {
        int total = Bn * Csrc * hw;
        chan_copy_kernel<<<(total + 255) / 256, 256, 0, stream>>>(src, dst, Bn, Csrc, Cdst, coff, hw);
    };
    auto kfsg = [&](const float* in, float* dst) {
        conv(in,    120, 121, nullptr, t128,  3,   H, Wd, 128, 3, 1, 1, 1);
        conv(t128,  122, 123, nullptr, t256a, 128, H, Wd, 256, 3, 1, 1, 1);
        conv(t256a, 124, 125, nullptr, t256b, 256, H, Wd, 256, 3, 1, 1, 1);
        conv(t256b, 126, 127, nullptr, dst,   256, H, Wd, 68,  3, 1, 1, 0);
    };
    auto fac = [&](const float* x, int base, int Cin, float* dst) {
        conv(x, base, base + 1, nullptr, fach, Cin, 32, 32, 128, 3, 1, 1, 1);
        resblock(fach, base + 2, 128, 32, 32);
        resblock(fach, base + 6, 128, 32, 32);
        conv(fach, base + 10, base + 11, nullptr, dst, 128, 32, 32, 3200, 1, 1, 0, 1);
    };

    const float* img_blur        = pf(161);
    const float* last_img_blur   = pf(162);
    const float* output_last_img = pf(163);
    const float* recovered = img_blur;
    const float* olf = pf(164);
    const float* flf = pf(165);

    for (int s = 0; s < 2; ++s) {
        // hm5 from current `recovered` (step>0: identical to stored hms[s-1])
        const float* hmsrc;
        if (s == 0) { kfsg(recovered, hm68); hmsrc = hm68; }
        else        hmsrc = out_hms + (size_t)(s - 1) * Bn * 68 * HW;
        hm_max_kernel<<<Bn * 68, 256, 0, stream>>>(hmsrc, mx, HW);
        { int total = Bn * HW;
          hm_merge5_kernel<<<(total + 255) / 256, 256, 0, stream>>>(hmsrc, mx, hm5, Bn, HW); }

        // merge = cat(img_blur, recovered, last_img_blur, output_last_img)
        copyCh(img_blur,        merge, 3, 12, 0, HW);
        copyCh(recovered,       merge, 3, 12, 3, HW);
        copyCh(last_img_blur,   merge, 3, 12, 6, HW);
        copyCh(output_last_img, merge, 3, 12, 9, HW);

        // c1 path
        conv(recovered, 0, 1, nullptr, c1buf, 3, H, Wd, 32, 3, 1, 1, 1);
        resblock(c1buf, 2, 32, H, Wd);
        resblock(c1buf, 6, 32, H, Wd);

        // kernel branch k1->k2->k3
        conv(merge, 40, 41, nullptr, k1buf, 12, H, Wd, 32, 3, 1, 1, 1);
        resblock(k1buf, 42, 32, H, Wd);
        resblock(k1buf, 46, 32, H, Wd);
        conv(k1buf, 50, 51, nullptr, k2buf, 32, H, Wd, 64, 3, 2, 1, 1);
        resblock(k2buf, 52, 64, 64, 64);
        resblock(k2buf, 56, 64, 64, 64);
        conv(k2buf, 60, 61, nullptr, k3buf, 64, 64, 64, 128, 3, 2, 1, 1);
        resblock(k3buf, 62, 128, 32, 32);
        resblock(k3buf, 66, 128, 32, 32);

        // FAC kernels
        fac(k3buf, 70, 128, kwarp);
        conv(kwarp, 82, 83, nullptr, kc4buf, 3200, 32, 32, 128, 1, 1, 0, 1);
        copyCh(k3buf,  cat256, 128, 256, 0,   HW4);
        copyCh(kc4buf, cat256, 128, 256, 128, HW4);
        fac(cat256, 84, 256, kdeb);

        // fusion (32 -> 160 -> attention-merge -> 32)
        conv(c1buf, 128, 129, nullptr, fused160, 32, H, Wd, 160, 3, 1, 1, 1);
        for (int i = 0; i < 7; ++i) resblock(fused160, 130 + 4 * i, 160, H, Wd);
        { int total = Bn * 32 * HW;
          fusion_apply_kernel<<<(total + 255) / 256, 256, 0, stream>>>(fused160, hm5, fused32, Bn, 32, HW); }

        // feature (stride-4) and encoder c2/c3
        float* featdst = (s == 1) ? out_feat : featbuf;
        conv(fused32, 30, 31, nullptr, featdst, 32, H, Wd, 64, 3, 4, 1, 1);
        resblock(featdst, 32, 64, 32, 32);
        resblock(featdst, 36, 64, 32, 32);

        conv(fused32, 10, 11, nullptr, c2buf, 32, H, Wd, 64, 3, 2, 1, 1);
        resblock(c2buf, 12, 64, 64, 64);
        resblock(c2buf, 16, 64, 64, 64);
        conv(c2buf, 20, 21, nullptr, c3buf, 64, 64, 64, 128, 3, 2, 1, 1);
        resblock(c3buf, 22, 128, 32, 32);
        resblock(c3buf, 26, 128, 32, 32);

        // per-pixel convs
        { int total = Bn * 128 * HW4;
          fac_conv_kernel<<<(total + 255) / 256, 256, 0, stream>>>(c3buf, kdeb, c3k, Bn, 128, 32, 32); }
        conv(olf, 96, 97, nullptr, olff, 256, 32, 32, 128, 3, 1, 1, 1);
        { int total = Bn * 128 * HW4;
          fac_conv_kernel<<<(total + 255) / 256, 256, 0, stream>>>(olff, kwarp, cakbuf, Bn, 128, 32, 32); }

        // compress_in (1x1, prelu)
        copyCh(c3k,    cat320, 128, 320, 0,   HW4);
        copyCh(cakbuf, cat320, 128, 320, 128, HW4);
        copyCh(flf,    cat320, 64,  320, 256, HW4);
        float* conv3dst = (s == 1) ? out_conv3 : conv3buf;
        conv(cat320, 158, 159, nullptr, conv3dst, 320, 32, 32, 256, 1, 1, 0, 2);

        // decoder
        { int total = Bn * 64 * HW2;
          upconv_kernel<<<(total + 255) / 256, 256, 0, stream>>>(conv3dst, pf(98), pf(99), up2buf, Bn, 256, 32, 32, 64); }
        resblock(up2buf, 100, 64, 64, 64);
        resblock(up2buf, 104, 64, 64, 64);
        { int total = Bn * 32 * HW;
          upconv_kernel<<<(total + 255) / 256, 256, 0, stream>>>(up2buf, pf(108), pf(109), up1buf, Bn, 64, 64, 64, 32); }
        resblock(up1buf, 110, 32, H, Wd);
        resblock(up1buf, 114, 32, H, Wd);

        // out_img = conv(up1) + recovered  (written straight into d_out)
        float* out_img = out_imgs + (size_t)s * Bn * 3 * HW;
        conv(up1buf, 118, 119, recovered, out_img, 32, H, Wd, 3, 3, 1, 1, 0);

        // hms[s] = HG2(out_img)
        kfsg(out_img, out_hms + (size_t)s * Bn * 68 * HW);

        recovered = out_img;
        olf = conv3dst;
        flf = featdst;
    }
}
